// CrossAttention_27530740367860
// MI455X (gfx1250) — compile-verified, matrix operand order checked by reading, not run on hardware
//
#include <hip/hip_runtime.h>
#include <hip/hip_bf16.h>

typedef __bf16 bf16;
typedef __attribute__((ext_vector_type(16))) __bf16 v16bf;
typedef __attribute__((ext_vector_type(8)))  __bf16 v8bf;
typedef __attribute__((ext_vector_type(8)))  float  v8f;

#define WMMA_BF16(a, b, c) \
    __builtin_amdgcn_wmma_f32_16x16x32_bf16(false, (a), false, (b), (short)0, (c), false, false)

// ---------------------------------------------------------------------------
// Fragment loader (CDNA5 WMMA 16x16x32 bf16 layouts, ISA 7.12.2)
// A frag: 16 rows x 32 K.  lane<16: row=lane, K {0..7, 16..23};
//                          lane>=16: row=lane-16, K {8..15, 24..31}.
// B frag is identical with "row" = output column (operand stored transposed).
// ---------------------------------------------------------------------------
__device__ __forceinline__ v16bf load_frag16(const bf16* base, int ld) {
    int lane = threadIdx.x & 31;
    int row  = lane & 15;
    int kh   = (lane >> 4) << 3;          // 0 or 8
    const bf16* p = base + (size_t)row * ld + kh;
    v16bf f;
    ((v8bf*)&f)[0] = *(const v8bf*)(p);        // K kh..kh+7
    ((v8bf*)&f)[1] = *(const v8bf*)(p + 16);   // K 16+kh..16+kh+7
    return f;
}

// ---------------------------------------------------------------------------
// 0) Weight convert + transpose: Wt[n][k] = (bf16)W[k][n]
// ---------------------------------------------------------------------------
__global__ void cvt_wt(const float* __restrict__ W, bf16* __restrict__ Wt,
                       int K, int N) {
    int idx = blockIdx.x * blockDim.x + threadIdx.x;
    if (idx < K * N) {
        int n = idx / K, k = idx - n * K;
        Wt[idx] = (bf16)W[(size_t)k * N + n];
    }
}

// ---------------------------------------------------------------------------
// 1) LayerNorm over channels, folded (n,c,s)->(n,s,c) transpose, bf16 out.
//    Block: 256 threads handle 64 rows (4 threads x 128 channels per row).
// ---------------------------------------------------------------------------
__global__ void ln_kernel(const float* __restrict__ X, const float* __restrict__ g,
                          const float* __restrict__ bta, bf16* __restrict__ out) {
    const int C = 512, S = 1024;
    int row0 = blockIdx.x << 6;
    int nb = row0 >> 10, s0 = row0 & 1023;
    int tid = threadIdx.x;
    int r = tid >> 2, part = tid & 3;
    const float* xp = X + (size_t)nb * C * S + (s0 + r);
    int c0 = part << 7;
    float sum = 0.f, ss = 0.f;
    for (int ch = c0; ch < c0 + 128; ++ch) {
        float v = xp[(size_t)ch << 10];
        sum += v; ss += v * v;
    }
    __shared__ float rs[64][4], rq[64][4];
    rs[r][part] = sum; rq[r][part] = ss;
    __syncthreads();
    sum = rs[r][0] + rs[r][1] + rs[r][2] + rs[r][3];
    ss  = rq[r][0] + rq[r][1] + rq[r][2] + rq[r][3];
    float mu   = sum * (1.f / 512.f);
    float var  = ss * (1.f / 512.f) - mu * mu;
    float rstd = rsqrtf(var + 1e-5f);
    bf16* op = out + (size_t)(row0 + r) * C;
    for (int ch = c0; ch < c0 + 128; ++ch) {
        float v = xp[(size_t)ch << 10];
        op[ch] = (bf16)((v - mu) * rstd * g[ch] + bta[ch]);
    }
}

// ---------------------------------------------------------------------------
// 2) GEMM (+bias) into the concatenated [q_proj | kv] bf16 tensor (stride 1536)
//    A: M x 512 bf16 row-major.  Wt: N x 512 bf16 (pre-transposed).
//    Block = 256 threads (8 waves); tile 64 rows x 128 cols; 16x16 per wave-tile.
// ---------------------------------------------------------------------------
__global__ void gemm_cat(const bf16* __restrict__ A, const bf16* __restrict__ Wt,
                         const float* __restrict__ bias, bf16* __restrict__ cat,
                         int colOff) {
    int row0    = blockIdx.x << 6;
    int colBase = blockIdx.y << 7;
    int wave = threadIdx.x >> 5, lane = threadIdx.x & 31;
    for (int t = wave; t < 32; t += 8) {
        int mt = t >> 3, nt = t & 7;
        const bf16* ab = A  + (size_t)(row0 + mt * 16) * 512;
        const bf16* wb = Wt + (size_t)(colBase + nt * 16) * 512;
        v8f acc = {0.f, 0.f, 0.f, 0.f, 0.f, 0.f, 0.f, 0.f};
#pragma unroll 4
        for (int kk = 0; kk < 512; kk += 32) {
            v16bf af = load_frag16(ab + kk, 512);
            v16bf bfg = load_frag16(wb + kk, 512);
            acc = WMMA_BF16(af, bfg, acc);
        }
        int ncol = colBase + nt * 16 + (lane & 15);
        float bv = bias[ncol];
        int rb = (lane < 16) ? 0 : 8;
#pragma unroll
        for (int i = 0; i < 8; ++i) {
            int row = row0 + mt * 16 + rb + i;
            cat[(size_t)row * 1536 + colOff + ncol] = (bf16)(acc[i] + bv);
        }
    }
}

// ---------------------------------------------------------------------------
// 2b) Transpose V slices out of cat: Vt[n][h][d (64)][s (1024)] bf16.
//     Makes every attention V B-fragment a contiguous 16-byte load.
// ---------------------------------------------------------------------------
__global__ void transpose_v(const bf16* __restrict__ Cat, bf16* __restrict__ Vt) {
    // total = 8 batches * 8 heads * 64 d * 1024 s = 4M elements
    int idx = blockIdx.x * blockDim.x + threadIdx.x;   // grid sized exactly
    int s   = idx & 1023;
    int t   = idx >> 10;          // (nb*8 + h)*64 + d
    int d   = t & 63;
    int nh  = t >> 6;             // nb*8 + h
    int h   = nh & 7;
    int nb  = nh >> 3;
    Vt[idx] = Cat[((size_t)nb * 1024 + s) * 1536 + h * 192 + 128 + d];
}

// ---------------------------------------------------------------------------
// 3) Flash attention. Grid (qtile=8, head=8, batch=8), 256 threads = 8 waves.
//    Wave owns 16 q rows; K loop in chunks of 32 keys; online softmax in f32.
//    Head slices follow the reference's concat-then-reshape: for head h,
//    q = cat[:, 192h .. +64), k = cat[:, 192h+64 ..), v = Vt[n][h].
// ---------------------------------------------------------------------------
__global__ void attn_kernel(const bf16* __restrict__ Cat, const bf16* __restrict__ Vt,
                            bf16* __restrict__ Ows) {
    const float LOG2E = 1.44269504f;
    int qt = blockIdx.x, h = blockIdx.y, nb = blockIdx.z;
    int wave = threadIdx.x >> 5, lane = threadIdx.x & 31;
    int q0 = qt * 128 + wave * 16;
    const bf16* catb  = Cat + (size_t)nb * 1024 * 1536;
    const bf16* qbase = catb + (size_t)q0 * 1536 + h * 192;
    const bf16* vtb   = Vt + ((size_t)(nb * 8 + h)) * 64 * 1024;  // [d][s]

    v16bf qa0 = load_frag16(qbase, 1536);        // K dims 0..31
    v16bf qa1 = load_frag16(qbase + 32, 1536);   // K dims 32..63

    float mrow[8], lrow[8], corr[8];
#pragma unroll
    for (int i = 0; i < 8; ++i) { mrow[i] = -1e30f; lrow[i] = 0.f; }
    v8f o0 = {0.f,0.f,0.f,0.f,0.f,0.f,0.f,0.f};
    v8f o1 = o0, o2 = o0, o3 = o0;

    __shared__ __align__(16) bf16 Pb[8][16][32];
    bf16 (*pb)[32] = Pb[wave];

    for (int kb = 0; kb < 1024; kb += 32) {
        const bf16* kp = catb + (size_t)kb * 1536 + h * 192 + 64;
        // S = q @ k^T over 32 key rows (two 16-col halves), K=64 via 2 WMMAs
        v16bf b00 = load_frag16(kp, 1536);
        v16bf b01 = load_frag16(kp + 32, 1536);
        v16bf b10 = load_frag16(kp + (size_t)16 * 1536, 1536);
        v16bf b11 = load_frag16(kp + (size_t)16 * 1536 + 32, 1536);
        v8f s0 = {0.f,0.f,0.f,0.f,0.f,0.f,0.f,0.f};
        v8f s1 = s0;
        s0 = WMMA_BF16(qa0, b00, s0);  s0 = WMMA_BF16(qa1, b01, s0);
        s1 = WMMA_BF16(qa0, b10, s1);  s1 = WMMA_BF16(qa1, b11, s1);

#pragma unroll
        for (int i = 0; i < 8; ++i) {
            float a0 = s0[i] * 0.125f, a1 = s1[i] * 0.125f;
            float rm = fmaxf(a0, a1);
            rm = fmaxf(rm, __shfl_xor(rm, 1));
            rm = fmaxf(rm, __shfl_xor(rm, 2));
            rm = fmaxf(rm, __shfl_xor(rm, 4));
            rm = fmaxf(rm, __shfl_xor(rm, 8));
            float mn = fmaxf(mrow[i], rm);
            float cr = exp2f((mrow[i] - mn) * LOG2E);
            float p0 = exp2f((a0 - mn) * LOG2E);
            float p1 = exp2f((a1 - mn) * LOG2E);
            float rsum = p0 + p1;
            rsum += __shfl_xor(rsum, 1);
            rsum += __shfl_xor(rsum, 2);
            rsum += __shfl_xor(rsum, 4);
            rsum += __shfl_xor(rsum, 8);
            lrow[i] = lrow[i] * cr + rsum;
            mrow[i] = mn; corr[i] = cr;
            int rr = i + ((lane < 16) ? 0 : 8);
            int cc = lane & 15;
            pb[rr][cc]      = (bf16)p0;   // restage P in A-fragment layout
            pb[rr][16 + cc] = (bf16)p1;
        }
        asm volatile("s_wait_dscnt 0" ::: "memory");   // cross-lane LDS visibility
        v16bf pf = load_frag16(&pb[0][0], 32);

#pragma unroll
        for (int i = 0; i < 8; ++i) {
            o0[i] *= corr[i]; o1[i] *= corr[i];
            o2[i] *= corr[i]; o3[i] *= corr[i];
        }
        // V B-frags from transposed Vt: contiguous 16B loads, ld = 1024
        const bf16* vb = vtb + kb;
        o0 = WMMA_BF16(pf, load_frag16(vb,                     1024), o0);
        o1 = WMMA_BF16(pf, load_frag16(vb + (size_t)16 * 1024, 1024), o1);
        o2 = WMMA_BF16(pf, load_frag16(vb + (size_t)32 * 1024, 1024), o2);
        o3 = WMMA_BF16(pf, load_frag16(vb + (size_t)48 * 1024, 1024), o3);
    }

    float inv[8];
#pragma unroll
    for (int i = 0; i < 8; ++i) inv[i] = 1.f / lrow[i];
    int cc = lane & 15, rb = (lane < 16) ? 0 : 8;
    bf16* ob = Ows + (size_t)(nb * 1024 + q0) * 512 + h * 64;
#pragma unroll
    for (int i = 0; i < 8; ++i) {
        int row = rb + i;
        ob[(size_t)row * 512 + cc]      = (bf16)(o0[i] * inv[i]);
        ob[(size_t)row * 512 + 16 + cc] = (bf16)(o1[i] * inv[i]);
        ob[(size_t)row * 512 + 32 + cc] = (bf16)(o2[i] * inv[i]);
        ob[(size_t)row * 512 + 48 + cc] = (bf16)(o3[i] * inv[i]);
    }
}

// ---------------------------------------------------------------------------
// 4) Output projection + bias, storing f32 into (n,c,h,w) layout.
// ---------------------------------------------------------------------------
__global__ void proj_kernel(const bf16* __restrict__ A, const bf16* __restrict__ Wt,
                            const float* __restrict__ bias, float* __restrict__ out) {
    int row0    = blockIdx.x << 6;
    int colBase = blockIdx.y << 7;
    int wave = threadIdx.x >> 5, lane = threadIdx.x & 31;
    for (int t = wave; t < 32; t += 8) {
        int mt = t >> 3, nt = t & 7;
        const bf16* ab = A  + (size_t)(row0 + mt * 16) * 512;
        const bf16* wb = Wt + (size_t)(colBase + nt * 16) * 512;
        v8f acc = {0.f, 0.f, 0.f, 0.f, 0.f, 0.f, 0.f, 0.f};
#pragma unroll 4
        for (int kk = 0; kk < 512; kk += 32) {
            v16bf af = load_frag16(ab + kk, 512);
            v16bf bfg = load_frag16(wb + kk, 512);
            acc = WMMA_BF16(af, bfg, acc);
        }
        int ncol = colBase + nt * 16 + (lane & 15);
        float bv = bias[ncol];
        int rb = (lane < 16) ? 0 : 8;
#pragma unroll
        for (int i = 0; i < 8; ++i) {
            int row  = row0 + mt * 16 + rb + i;
            int nb   = row >> 10;
            int spos = row & 1023;
            out[((size_t)nb * 512 + ncol) * 1024 + spos] = acc[i] + bv;
        }
    }
}

// ---------------------------------------------------------------------------
extern "C" void kernel_launch(void* const* d_in, const int* in_sizes, int n_in,
                              void* d_out, int out_size, void* d_ws, size_t ws_size,
                              hipStream_t stream) {
    const float* x       = (const float*)d_in[0];
    const float* query_x = (const float*)d_in[1];
    const float* q_ln_g  = (const float*)d_in[2];
    const float* q_ln_b  = (const float*)d_in[3];
    const float* q_w     = (const float*)d_in[4];
    const float* q_b     = (const float*)d_in[5];
    const float* kv_ln_g = (const float*)d_in[6];
    const float* kv_ln_b = (const float*)d_in[7];
    const float* kv_w    = (const float*)d_in[8];
    const float* kv_b    = (const float*)d_in[9];
    const float* proj_w  = (const float*)d_in[10];
    const float* proj_b  = (const float*)d_in[11];
    float* out = (float*)d_out;

    // workspace layout (bf16)
    char* ws = (char*)d_ws;
    bf16* qwt  = (bf16*)(ws);                         // 512x512   = 0.5 MB
    bf16* kvwt = (bf16*)(ws + (512u * 512u * 2u));    // 1024x512  = 1.0 MB
    bf16* pwt  = (bf16*)(ws + (512u * 512u * 2u) + (1024u * 512u * 2u));
    char* p    = ws + 2u * 1024u * 1024u;             // aligned region
    bf16* lnq  = (bf16*)(p);                          // 8192x512  (8 MB)
    bf16* lnx  = (bf16*)(p + 8ull * 1024 * 1024);     // 8192x512  (8 MB)
    bf16* cat  = (bf16*)(p + 16ull * 1024 * 1024);    // 8192x1536 (24 MB)
    bf16* ows  = (bf16*)(p + 40ull * 1024 * 1024);    // 8192x512  (8 MB)
    bf16* vt   = (bf16*)(p + 48ull * 1024 * 1024);    // 8x8x64x1024 (8 MB)

    // 0) weights -> bf16, transposed
    cvt_wt<<<(512 * 512 + 255) / 256, 256, 0, stream>>>(q_w, qwt, 512, 512);
    cvt_wt<<<(512 * 1024 + 255) / 256, 256, 0, stream>>>(kv_w, kvwt, 512, 1024);
    cvt_wt<<<(512 * 512 + 255) / 256, 256, 0, stream>>>(proj_w, pwt, 512, 512);

    // 1) LayerNorms (fold NCHW->NSC transpose)
    ln_kernel<<<dim3(8192 / 64), 256, 0, stream>>>(query_x, q_ln_g, q_ln_b, lnq);
    ln_kernel<<<dim3(8192 / 64), 256, 0, stream>>>(x, kv_ln_g, kv_ln_b, lnx);

    // 2) projections into concatenated [q_proj | kv]
    gemm_cat<<<dim3(128, 4), 256, 0, stream>>>(lnq, qwt, q_b, cat, 0);
    gemm_cat<<<dim3(128, 8), 256, 0, stream>>>(lnx, kvwt, kv_b, cat, 512);

    // 2b) V transpose for contiguous B-fragments
    transpose_v<<<(8 * 8 * 64 * 1024) / 256, 256, 0, stream>>>(cat, vt);

    // 3) flash attention per (qtile, head, batch)
    attn_kernel<<<dim3(8, 8, 8), 256, 0, stream>>>(cat, vt, ows);

    // 4) output projection -> (n,c,h,w) f32
    proj_kernel<<<dim3(128, 4), 256, 0, stream>>>(ows, pwt, proj_b, out);
}
